// TrinetBSRNN_13615046328329
// MI455X (gfx1250) — compile-verified, matrix operand order checked by reading, not run on hardware
//
#include <hip/hip_runtime.h>
#include <hip/hip_bf16.h>

typedef __attribute__((ext_vector_type(16))) __bf16 v16bf;
typedef __attribute__((ext_vector_type(8)))  float  v8f;
typedef __attribute__((ext_vector_type(4)))  unsigned uint4v;
typedef __attribute__((ext_vector_type(8)))  int      int8v;
typedef __attribute__((ext_vector_type(4)))  int      int4v;

#define DEVFN __device__ __forceinline__

// ---------------- geometry (fixed by the reference) ----------------
constexpr int BATCH = 4, CIN = 512, COUT = 512, CH = 256, D2 = 256, D1 = 64;
constexpr int RTOT  = BATCH * D2 * D1;   // 65536 spatial positions
constexpr int NSP   = D2 * D1;           // 16384 spatial per batch image

// ---------------- helpers ----------------
union FragU { v16bf v; uint4 q[2]; };

DEVFN float b2f(__hip_bfloat16 h) { return __bfloat162float(h); }

DEVFN float wred_sum(float v) {
#pragma unroll
  for (int o = 16; o > 0; o >>= 1) v += __shfl_xor(v, o, 32);
  return v;
}
DEVFN float wred_max(float v) {
#pragma unroll
  for (int o = 16; o > 0; o >>= 1) v = fmaxf(v, __shfl_xor(v, o, 32));
  return v;
}

// low 32 bits of a flat shared-memory address = LDS byte offset (ISA 10.2)
DEVFN unsigned lds_off(const void* p) { return (unsigned)(unsigned long long)p; }

// =====================================================================
// TDM: issue a 2D bf16 tile load (tile_dim0=32 elems, tile_dim1=64 rows)
// into LDS with hardware padding matching LDSS=40 (after every 16 DWORDs
// insert 4 DWORDs).  Rows beyond `rows` are zero-filled by TDM OOB logic.
// D# fields per CDNA5 ISA 8.3/8.4; 6-arg builtin (this toolchain):
//   (uint32x4 g0, int32x8 g1, int32x4 g2, int32x4 g3, int32x8, i32 cpol)
// groups 2/3 zero => tensor up to 2D.
// =====================================================================
DEVFN void tdm_load_tile(unsigned lds, const __hip_bfloat16* g, int rows,
                         long long strideElems) {
  unsigned long long ga = (unsigned long long)g;
  unsigned td1 = (unsigned)(rows < 1 ? 1 : rows);
  unsigned long long st = (unsigned long long)strideElems;
  uint4v g0 = { 1u,                                   // count=1, user load
                lds,                                  // lds_addr
                (unsigned)ga,                         // global_addr[31:0]
                (unsigned)((ga >> 32) & 0x1ffffffu) | (2u << 30) }; // hi + type=2
  int8v g1 = {
    (int)((1u << 16)      // data_size = 2 bytes
        | (1u << 20)      // pad_enable
        | (3u << 22)      // pad_interval: 16 DWORDs
        | (3u << 25)),    // pad_amount : 4 DWORDs
    (int)((32u & 0xffffu) << 16),                 // tensor_dim0 lo16 (=32)
    (int)((32u >> 16) | ((td1 & 0xffffu) << 16)), // tensor_dim0 hi | tensor_dim1 lo
    (int)((td1 >> 16) | (32u << 16)),             // tensor_dim1 hi | tile_dim0=32
    (int)(64u),                                   // tile_dim1=64, tile_dim2=0
    (int)(st & 0xffffffffu),                      // tensor_dim0_stride lo32
    (int)((st >> 32) & 0xffffu),                  // stride hi16 | dim1_stride lo=0
    0 };
  int4v z4 = { 0, 0, 0, 0 };
  int8v z8 = { 0, 0, 0, 0, 0, 0, 0, 0 };
  __builtin_amdgcn_tensor_load_to_lds(g0, g1, z4, z4, z8, 0);
}

// =====================================================================
// Batched bf16 WMMA GEMM:  C[M,N] = A[M,K] * Bt[N,K] (+bias, epilogue)
//   A: row-major bf16 [M, lda]
//   B: if !bKmajor: row-major bf16 [N, ldb]  -> staged by TDM (DMA)
//      if  bKmajor: bf16 [K, ldb], column n  -> cooperative transposing stage
//   K multiple of 32 (all uses).  OOB rows: TDM zero-fills; cooperative
//   path clamps (outputs masked in epilogue either way).
//   epi: 0 = f32 out, 1 = bf16 out, 2 = stage0 (bias+prelu+PE -> f32+bf16),
//        4 = final NCHW scatter to d_out
// =====================================================================
struct GemmP {
  const __hip_bfloat16* A;
  const __hip_bfloat16* B;
  const float* bias;
  float* Cf; __hip_bfloat16* Cb;
  long long lda, ldb, ldc;
  long long sA, sB, sC;          // batch strides in elements
  int M, N, K;
  int bKmajor, epi;
  const float* p0; const float* p1;   // device scalars (alpha, pe_scale)
};

constexpr int LDSS = 40;   // padded LDS row stride in bf16 (32 + 8)

__launch_bounds__(128)
__global__ void k_wmma_gemm(GemmP p) {
  __shared__ __bf16 As[2][64 * LDSS];
  __shared__ __bf16 Bs[2][64 * LDSS];
  const int n0 = blockIdx.x * 64;
  const int m0 = blockIdx.y * 64;
  const long long bz = blockIdx.z;
  const int t    = threadIdx.x;
  const int lane = t & 31, wave = t >> 5;
  const int wm = wave >> 1, wn = wave & 1;   // 2x2 waves over 64x64 tile
  const int fr = lane & 15, fh = lane >> 4;  // fragment row / half

  v8f acc[2][2] = {};

  // tile base pointers (uniform)
  const __hip_bfloat16* aTile = p.A + bz * p.sA + (long long)m0 * p.lda;
  const __hip_bfloat16* bTile = p.B + bz * p.sB + (long long)n0 * p.ldb;

  // cooperative staging assignments (k-major / fallback path)
  const int srow = t >> 1;
  const int sks  = (t & 1) * 16;
  const int agm  = min(m0 + srow, p.M - 1);
  const __hip_bfloat16* aSrc = p.A + bz * p.sA + (long long)agm * p.lda + sks;
  const int kp = t >> 3;          // 16 k-pairs
  const int ng = (t & 7) * 8;     // 8 n each
  const __hip_bfloat16* btSrc = p.B + bz * p.sB + (long long)(2 * kp) * p.ldb + n0 + ng;

  auto issue = [&](int kk, int buf) {
    if (!p.bKmajor) {
      if (wave == 0) {   // one wave drives the Tensor Data Mover for A and B
        tdm_load_tile(lds_off(&As[buf][0]), aTile + kk, p.M - m0, p.lda);
        tdm_load_tile(lds_off(&Bs[buf][0]), bTile + kk, p.N - n0, p.ldb);
      }
    } else {
      // A tile cooperatively
      const __hip_bfloat16* s = aSrc + kk;
      __bf16* Ad = As[buf];
      *(uint4*)(Ad + srow * LDSS + sks)     = *(const uint4*)s;
      *(uint4*)(Ad + srow * LDSS + sks + 8) = *(const uint4*)(s + 8);
      if (kk + 32 < p.K) __builtin_prefetch((const void*)(s + 32), 0, 1);
      // B tile: read two contiguous rows (coalesced), write packed k-pairs
      const __hip_bfloat16* s0 = btSrc + (long long)kk * p.ldb;
      const __hip_bfloat16* s1 = s0 + p.ldb;
      union { uint4 q; unsigned short u[8]; } r0, r1;
      r0.q = *(const uint4*)s0;
      r1.q = *(const uint4*)s1;
      __bf16* Bd = Bs[buf];
#pragma unroll
      for (int j = 0; j < 8; ++j) {
        unsigned pk = (unsigned)r0.u[j] | ((unsigned)r1.u[j] << 16);
        *(unsigned*)(Bd + (ng + j) * LDSS + 2 * kp) = pk;
      }
      if (kk + 32 < p.K) __builtin_prefetch((const void*)(s0 + 32ll * p.ldb), 0, 1);
    }
  };

  issue(0, 0);

  for (int k0 = 0; k0 < p.K; k0 += 32) {
    const int cur = (k0 >> 5) & 1;
    // unconditional: TENSORcnt is per-wave and 0 on non-issuing waves
    __builtin_amdgcn_s_wait_tensorcnt(0);
    __syncthreads();                       // staged tile `cur` now visible
    if (k0 + 32 < p.K) issue(k0 + 32, cur ^ 1);

    // fragments; lane(r,h): K = h*8+0..7 and 16+h*8+0..7
    FragU af[2], bfr[2];
    const __bf16* Ab = As[cur];
    const __bf16* Bb = Bs[cur];
#pragma unroll
    for (int mi = 0; mi < 2; ++mi) {
      const __bf16* s = Ab + (wm * 32 + mi * 16 + fr) * LDSS + fh * 8;
      af[mi].q[0] = *(const uint4*)s;
      af[mi].q[1] = *(const uint4*)(s + 16);
    }
#pragma unroll
    for (int ni = 0; ni < 2; ++ni) {
      const __bf16* s = Bb + (wn * 32 + ni * 16 + fr) * LDSS + fh * 8;
      bfr[ni].q[0] = *(const uint4*)s;
      bfr[ni].q[1] = *(const uint4*)(s + 16);
    }
#pragma unroll
    for (int mi = 0; mi < 2; ++mi)
#pragma unroll
      for (int ni = 0; ni < 2; ++ni)
        acc[mi][ni] = __builtin_amdgcn_wmma_f32_16x16x32_bf16(
            false, af[mi].v, false, bfr[ni].v, (short)0, acc[mi][ni], false, false);
  }

  // ---- epilogue ----
  const float e0 = p.p0 ? *p.p0 : 0.f;
  const float e1 = p.p1 ? *p.p1 : 0.f;
#pragma unroll
  for (int ni = 0; ni < 2; ++ni) {
    const int gc = n0 + wn * 32 + ni * 16 + fr;
    if (gc >= p.N) continue;
    const float bb = p.bias ? p.bias[gc] : 0.f;
#pragma unroll
    for (int mi = 0; mi < 2; ++mi)
#pragma unroll
      for (int i = 0; i < 8; ++i) {
        const int gr = m0 + wm * 32 + mi * 16 + i + 8 * fh;
        if (gr >= p.M) continue;
        float v = acc[mi][ni][i] + bb;
        const long long co = bz * p.sC + (long long)gr * p.ldc + gc;
        if (p.epi == 0) {
          p.Cf[co] = v;
        } else if (p.epi == 1) {
          p.Cb[co] = __float2bfloat16(v);
        } else if (p.epi == 2) {
          // stage0: prelu(a_in) + pe_scale*(tpe[c,h] + fpe[c,w])
          v = v >= 0.f ? v : e0 * v;
          const int ww = gr & (D1 - 1);
          const int hh = (gr >> 6) & (D2 - 1);
          const float dv = __expf(-0.0359778925f * (float)(2 * (gc >> 1)));
          const float pe = (gc & 1) ? (__cosf(hh * dv) + __cosf(ww * dv))
                                    : (__sinf(hh * dv) + __sinf(ww * dv));
          v += e1 * pe;
          p.Cf[co] = v;
          p.Cb[co] = __float2bfloat16(v);
        } else { // 4: scatter to NCHW output
          const int ww = gr & (D1 - 1);
          const int hh = (gr >> 6) & (D2 - 1);
          const int bb2 = gr >> 14;
          p.Cf[(((long long)bb2 * COUT + gc) * D2 + hh) * D1 + ww] = v;
        }
      }
  }
}

// =====================================================================
// Elementwise / reduction kernels
// =====================================================================
__launch_bounds__(256)
__global__ void k_f2b(const float* __restrict__ a, __hip_bfloat16* __restrict__ o, long long n) {
  for (long long i = blockIdx.x * 256ll + threadIdx.x; i < n; i += (long long)gridDim.x * 256)
    o[i] = __float2bfloat16(a[i]);
}

__launch_bounds__(256)
__global__ void k_nchw2nhwc(const float* __restrict__ x, __hip_bfloat16* __restrict__ o, long long n) {
  for (long long i = blockIdx.x * 256ll + threadIdx.x; i < n; i += (long long)gridDim.x * 256) {
    int c = (int)(i & (CIN - 1));
    long long m = i >> 9;                 // CIN = 512
    int b = (int)(m >> 14);
    int pos = (int)(m & (NSP - 1));
    o[i] = __float2bfloat16(x[((long long)b * CIN + c) * NSP + pos]);
  }
}

__launch_bounds__(256)
__global__ void k_row2col(const __hip_bfloat16* __restrict__ h, __hip_bfloat16* __restrict__ o, long long n) {
  // [b, hh, ww, c] -> [b, ww, hh, c]
  for (long long i = blockIdx.x * 256ll + threadIdx.x; i < n; i += (long long)gridDim.x * 256) {
    int c = (int)(i & (CH - 1));
    long long m = i >> 8;
    int b = (int)(m >> 14);
    int pos = (int)(m & (NSP - 1));
    int hh = pos >> 6, ww = pos & 63;
    o[((((long long)b * D1 + ww) * D2 + hh) << 8) + c] = h[i];
  }
}

__launch_bounds__(256)
__global__ void k_im2col(const __hip_bfloat16* __restrict__ h, __hip_bfloat16* __restrict__ o,
                         int T, long long n) {
  // out[(seq*Td+s), j*C+c] = h[seq, 2s+j-1, c] (zero pad); matches dswT [O,3,I]
  int TD = T >> 1;
  for (long long i = blockIdx.x * 256ll + threadIdx.x; i < n; i += (long long)gridDim.x * 256) {
    int c = (int)(i & (CH - 1));
    long long r = i >> 8;
    int j = (int)(r % 3);
    long long r2 = r / 3;
    int s   = (int)(r2 % TD);
    long long seq = r2 / TD;
    int t = 2 * s + j - 1;
    __hip_bfloat16 v = __float2bfloat16(0.f);
    if (t >= 0 && t < T) v = h[((seq * T + t) << 8) + c];
    o[i] = v;
  }
}

__launch_bounds__(256)
__global__ void k_prep_dsw(const float* __restrict__ w, __hip_bfloat16* __restrict__ o, long long n) {
  // [O, I, 3] -> [O, 3, I]
  for (long long i = blockIdx.x * 256ll + threadIdx.x; i < n; i += (long long)gridDim.x * 256) {
    int ii = (int)(i & (CH - 1));
    long long r = i >> 8;
    int j  = (int)(r % 3);
    int oo = (int)(r / 3);
    o[i] = __float2bfloat16(w[((long long)oo * CH + ii) * 3 + j]);
  }
}

__launch_bounds__(256)
__global__ void k_rownorm(const float* __restrict__ in, __hip_bfloat16* __restrict__ out, int rows) {
  int wave = threadIdx.x >> 5, lane = threadIdx.x & 31;
  int row = blockIdx.x * 8 + wave;
  if (row >= rows) return;
  const float* src = in + (long long)row * CH;
  float v[8]; float ss = 0.f;
#pragma unroll
  for (int k = 0; k < 8; ++k) { v[k] = src[lane + 32 * k]; ss += v[k] * v[k]; }
  ss = wred_sum(ss);
  float inv = 1.f / (sqrtf(ss) + 1e-8f);
#pragma unroll
  for (int k = 0; k < 8; ++k)
    out[(long long)row * CH + lane + 32 * k] = __float2bfloat16(v[k] * inv);
}

__launch_bounds__(256)
__global__ void k_softmax(const float* __restrict__ in, __hip_bfloat16* __restrict__ out,
                          int rows, int S, int mode, float sconst, const float* sptr) {
  int wave = threadIdx.x >> 5, lane = threadIdx.x & 31;
  int row = blockIdx.x * 8 + wave;
  if (row >= rows) return;
  float sc = (mode == 1) ? (1.f / fmaxf(*sptr, 0.01f)) : sconst;
  const float* src = in + (long long)row * S;
  int KM = (S + 31) >> 5;
  float z[8];
  float mx = -1e30f;
  for (int k = 0; k < KM; ++k) {
    int c = lane + 32 * k;
    z[k] = (c < S) ? src[c] * sc : -1e30f;
    mx = fmaxf(mx, z[k]);
  }
  mx = wred_max(mx);
  float sum = 0.f;
  for (int k = 0; k < KM; ++k) { z[k] = __expf(z[k] - mx); sum += z[k]; }
  sum = wred_sum(sum);
  float inv = 1.f / sum;
  for (int k = 0; k < KM; ++k) {
    int c = lane + 32 * k;
    if (c < S) out[(long long)row * S + c] = __float2bfloat16(z[k] * inv);
  }
}

__launch_bounds__(256)
__global__ void k_layernorm(const float* __restrict__ in, const float* __restrict__ g,
                            const float* __restrict__ b, __hip_bfloat16* __restrict__ out, int rows) {
  int wave = threadIdx.x >> 5, lane = threadIdx.x & 31;
  int row = blockIdx.x * 8 + wave;
  if (row >= rows) return;
  const float* src = in + (long long)row * CH;
  float v[8]; float s = 0.f;
#pragma unroll
  for (int k = 0; k < 8; ++k) { v[k] = src[lane + 32 * k]; s += v[k]; }
  s = wred_sum(s);
  float mean = s * (1.f / CH);
  float q = 0.f;
#pragma unroll
  for (int k = 0; k < 8; ++k) { float d = v[k] - mean; q += d * d; }
  q = wred_sum(q);
  float rstd = rsqrtf(q * (1.f / CH) + 1e-5f);
#pragma unroll
  for (int k = 0; k < 8; ++k) {
    int c = lane + 32 * k;
    out[(long long)row * CH + c] = __float2bfloat16((v[k] - mean) * rstd * g[c] + b[c]);
  }
}

__launch_bounds__(256)
__global__ void k_gate(const __hip_bfloat16* __restrict__ oc, const __hip_bfloat16* __restrict__ od,
                       const __hip_bfloat16* __restrict__ os, const float* __restrict__ W,
                       const float* __restrict__ gb, float* __restrict__ out, int rows) {
  int wave = threadIdx.x >> 5, lane = threadIdx.x & 31;
  int row = blockIdx.x * 8 + wave;
  if (row >= rows) return;
  long long base = (long long)row * CH;
  float vc[8], vd[8], vs[8];
  float g0 = 0.f, g1 = 0.f, g2 = 0.f;
#pragma unroll
  for (int k = 0; k < 8; ++k) {
    int c = lane + 32 * k;
    vc[k] = b2f(oc[base + c]); vd[k] = b2f(od[base + c]); vs[k] = b2f(os[base + c]);
    g0 += vc[k] * W[c] + vd[k] * W[256 + c] + vs[k] * W[512 + c];
    g1 += vc[k] * W[768 + c] + vd[k] * W[1024 + c] + vs[k] * W[1280 + c];
    g2 += vc[k] * W[1536 + c] + vd[k] * W[1792 + c] + vs[k] * W[2048 + c];
  }
  g0 = wred_sum(g0) + gb[0];
  g1 = wred_sum(g1) + gb[1];
  g2 = wred_sum(g2) + gb[2];
  float m = fmaxf(g0, fmaxf(g1, g2));
  float e0 = __expf(g0 - m), e1 = __expf(g1 - m), e2 = __expf(g2 - m);
  float inv = 1.f / (e0 + e1 + e2);
  e0 *= inv; e1 *= inv; e2 *= inv;
#pragma unroll
  for (int k = 0; k < 8; ++k) {
    int c = lane + 32 * k;
    out[base + c] = e0 * vc[k] + e1 * vd[k] + e2 * vs[k];
  }
}

__launch_bounds__(256)
__global__ void k_instat(const float* __restrict__ in, float* __restrict__ stats) {
  // one block per (b, c); in layout [b, p(NSP), c(CH)]
  int bc = blockIdx.x;
  int b = bc >> 8, c = bc & (CH - 1);
  const float* base = in + (long long)b * NSP * CH + c;
  float s = 0.f, q = 0.f;
  for (int p = threadIdx.x; p < NSP; p += 256) {
    float v = base[(long long)p * CH];
    s += v; q += v * v;
  }
  __shared__ float sh[16];
  s = wred_sum(s); q = wred_sum(q);
  int wave = threadIdx.x >> 5, lane = threadIdx.x & 31;
  if (lane == 0) { sh[wave] = s; sh[8 + wave] = q; }
  __syncthreads();
  if (threadIdx.x == 0) {
    float S = 0.f, Q = 0.f;
    for (int w = 0; w < 8; ++w) { S += sh[w]; Q += sh[8 + w]; }
    float mean = S * (1.f / NSP);
    float var = Q * (1.f / NSP) - mean * mean;
    stats[bc * 2] = mean;
    stats[bc * 2 + 1] = rsqrtf(var + 1e-5f);
  }
}

__launch_bounds__(256)
__global__ void k_combine(const float* __restrict__ h, const float* __restrict__ rowo,
                          const float* __restrict__ colo,
                          const float* __restrict__ stR, const float* __restrict__ stC,
                          const float* __restrict__ rg, const float* __restrict__ rb,
                          const float* __restrict__ cg, const float* __restrict__ cb,
                          const float* k1p, const float* k2p, const float* ap,
                          __hip_bfloat16* __restrict__ out, long long n) {
  float k1 = *k1p, k2 = *k2p, al = *ap;
  for (long long i = blockIdx.x * 256ll + threadIdx.x; i < n; i += (long long)gridDim.x * 256) {
    int c = (int)(i & (CH - 1));
    long long m = i >> 8;
    int b = (int)(m >> 14);
    int pos = (int)(m & (NSP - 1));
    int hh = pos >> 6, ww = pos & 63;
    int sc_i = (b << 8) + c;
    float rv = (rowo[i] - stR[sc_i * 2]) * stR[sc_i * 2 + 1] * rg[c] + rb[c];
    long long ci = ((((long long)b * D1 + ww) * D2 + hh) << 8) + c;
    float cv = (colo[ci] - stC[sc_i * 2]) * stC[sc_i * 2 + 1] * cg[c] + cb[c];
    float v = h[i] + k1 * rv + k2 * cv;
    v = v >= 0.f ? v : al * v;
    out[i] = __float2bfloat16(v);
  }
}

// =====================================================================
// Host side
// =====================================================================
static void gemm(hipStream_t st,
                 const __hip_bfloat16* A, long long lda, long long sA,
                 const __hip_bfloat16* B, bool bK, long long ldb, long long sB,
                 const float* bias, float* Cf, __hip_bfloat16* Cb,
                 long long ldc, long long sC,
                 int M, int N, int K, int batch, int epi,
                 const float* p0 = nullptr, const float* p1 = nullptr) {
  GemmP p{A, B, bias, Cf, Cb, lda, ldb, ldc, sA, sB, sC, M, N, K,
          bK ? 1 : 0, epi, p0, p1};
  dim3 g((N + 63) / 64, (M + 63) / 64, batch);
  k_wmma_gemm<<<g, 128, 0, st>>>(p);
}

static inline int gsblocks(long long n) {
  long long b = (n + 255) / 256;
  return (int)(b > 8192 ? 8192 : b);
}

// mrha param slot order (dict insertion order in _mrha_params)
enum { M_DSW = 0, M_DSB, M_QCW, M_QCB, M_KCW, M_KCB, M_VCW, M_VCB,
       M_QDW, M_QDB, M_KDW, M_KDB, M_VDW, M_VDB,
       M_QSW, M_QSB, M_KSW, M_KSB, M_VSW, M_VSB,
       M_LNCG, M_LNCB, M_LNDG, M_LNDB, M_LNSG, M_LNSB,
       M_COST, M_GATEW, M_GATEB, M_COUNT };

struct Mrha {
  const float* p[M_COUNT];
  const __hip_bfloat16* wb[9];   // bf16 qc,kc,vc,qd,kd,vd,qs,ks,vs
  __hip_bfloat16* dswT;
};

struct Ws {
  __hip_bfloat16 *xT, *hb, *hcol, *ic, *xd, *qb, *kb, *vb, *pb, *oc, *od, *os, *comb;
  __hip_bfloat16 *dswT_r, *dswT_c, *in_wb, *out_wb;
  float *hf, *rowo, *colo, *tmpf, *scores, *statsR, *statsC;
};

static void run_mrha(hipStream_t st, const Mrha& m, const __hip_bfloat16* hin,
                     float* outp, int NSEQ, int T, const Ws& w) {
  const int TD = T / 2;
  const int RT = NSEQ * T, RD = NSEQ * TD;
  const float scl = 0.0625f;   // 1/sqrt(256)

  // downsample conv as im2col GEMM
  {
    long long n = (long long)RD * 3 * CH;
    k_im2col<<<gsblocks(n), 256, 0, st>>>(hin, w.ic, T, n);
  }
  gemm(st, w.ic, 3 * CH, 0, m.dswT, false, 3 * CH, 0, m.p[M_DSB],
       nullptr, w.xd, CH, 0, RD, CH, 3 * CH, 1, 1);

  auto lin = [&](const __hip_bfloat16* X, int rows, const __hip_bfloat16* Wb, const float* bias,
                 __hip_bfloat16* outB, float* outF, int epi) {
    gemm(st, X, CH, 0, Wb, false, CH, 0, bias, outF, outB, CH, 0, rows, CH, CH, 1, epi);
  };
  auto attn = [&](const __hip_bfloat16* Q, const __hip_bfloat16* Kb, const __hip_bfloat16* Vb,
                  int S, int smode, const float* sptr,
                  const float* lg, const float* lb, __hip_bfloat16* outB) {
    // scores[t,s] = Q[t,:]·K[s,:]   (batched over sequences, TDM staging)
    gemm(st, Q, CH, (long long)T * CH, Kb, false, CH, (long long)S * CH,
         nullptr, w.scores, nullptr, S, (long long)T * S, T, S, CH, NSEQ, 0);
    k_softmax<<<(RT + 7) / 8, 256, 0, st>>>(w.scores, w.pb, RT, S, smode, scl, sptr);
    // out[t,c] = P[t,:]·V[:,c]   (V k-major, transposing cooperative staging)
    gemm(st, w.pb, S, (long long)T * S, Vb, true, CH, (long long)S * CH,
         nullptr, w.tmpf, nullptr, CH, (long long)T * CH, T, CH, S, NSEQ, 0);
    k_layernorm<<<(RT + 7) / 8, 256, 0, st>>>(w.tmpf, lg, lb, outB, RT);
  };

  // cross-resolution branch
  lin(hin, RT, m.wb[0], m.p[M_QCB], w.qb, nullptr, 1);
  lin(w.xd, RD, m.wb[1], m.p[M_KCB], w.kb, nullptr, 1);
  lin(w.xd, RD, m.wb[2], m.p[M_VCB], w.vb, nullptr, 1);
  attn(w.qb, w.kb, w.vb, TD, 0, nullptr, m.p[M_LNCG], m.p[M_LNCB], w.oc);

  // dot-product branch
  lin(hin, RT, m.wb[3], m.p[M_QDB], w.qb, nullptr, 1);
  lin(hin, RT, m.wb[4], m.p[M_KDB], w.kb, nullptr, 1);
  lin(hin, RT, m.wb[5], m.p[M_VDB], w.vb, nullptr, 1);
  attn(w.qb, w.kb, w.vb, T, 0, nullptr, m.p[M_LNDG], m.p[M_LNDB], w.od);

  // cosine branch (normalize q,k; temperature softmax)
  lin(hin, RT, m.wb[6], m.p[M_QSB], nullptr, w.tmpf, 0);
  k_rownorm<<<(RT + 7) / 8, 256, 0, st>>>(w.tmpf, w.qb, RT);
  lin(hin, RT, m.wb[7], m.p[M_KSB], nullptr, w.tmpf, 0);
  k_rownorm<<<(RT + 7) / 8, 256, 0, st>>>(w.tmpf, w.kb, RT);
  lin(hin, RT, m.wb[8], m.p[M_VSB], w.vb, nullptr, 1);
  attn(w.qb, w.kb, w.vb, T, 1, m.p[M_COST], m.p[M_LNSG], m.p[M_LNSB], w.os);

  // gated fusion of 3 branches
  k_gate<<<(RT + 7) / 8, 256, 0, st>>>(w.oc, w.od, w.os, m.p[M_GATEW], m.p[M_GATEB], outp, RT);
}

extern "C" void kernel_launch(void* const* d_in, const int* in_sizes, int n_in,
                              void* d_out, int out_size, void* d_ws, size_t ws_size,
                              hipStream_t stream) {
  (void)in_sizes; (void)n_in; (void)out_size; (void)ws_size;

  // flat input mapping (setup_inputs dict insertion order, recursive)
  const float* x    = (const float*)d_in[0];
  const float* in_w = (const float*)d_in[1];
  const float* in_b = (const float*)d_in[2];
  const float* a_in = (const float*)d_in[3];
  Mrha rowm, colm;
  for (int i = 0; i < M_COUNT; ++i) rowm.p[i] = (const float*)d_in[4 + i];
  for (int i = 0; i < M_COUNT; ++i) colm.p[i] = (const float*)d_in[4 + M_COUNT + i];
  int base = 4 + 2 * M_COUNT;                 // 62
  const float* rn_g  = (const float*)d_in[base + 0];
  const float* rn_b  = (const float*)d_in[base + 1];
  const float* cn_g  = (const float*)d_in[base + 2];
  const float* cn_b  = (const float*)d_in[base + 3];
  const float* k1    = (const float*)d_in[base + 4];
  const float* k2    = (const float*)d_in[base + 5];
  const float* pe    = (const float*)d_in[base + 6];
  const float* a_out = (const float*)d_in[base + 7];
  const float* out_w = (const float*)d_in[base + 8];
  const float* out_b = (const float*)d_in[base + 9];

  // workspace carve
  size_t off = 0;
  auto alloc = [&](size_t bytes) {
    void* p = (char*)d_ws + off;
    off = (off + bytes + 255) & ~(size_t)255;
    return p;
  };
  Ws w;
  w.xT     = (__hip_bfloat16*)alloc((size_t)RTOT * CIN * 2);
  w.hf     = (float*)alloc((size_t)RTOT * CH * 4);
  w.hb     = (__hip_bfloat16*)alloc((size_t)RTOT * CH * 2);
  w.hcol   = (__hip_bfloat16*)alloc((size_t)RTOT * CH * 2);
  w.rowo   = (float*)alloc((size_t)RTOT * CH * 4);
  w.colo   = (float*)alloc((size_t)RTOT * CH * 4);
  w.dswT_r = (__hip_bfloat16*)alloc((size_t)CH * 3 * CH * 2);
  w.dswT_c = (__hip_bfloat16*)alloc((size_t)CH * 3 * CH * 2);
  w.in_wb  = (__hip_bfloat16*)alloc((size_t)CH * CIN * 2);
  w.out_wb = (__hip_bfloat16*)alloc((size_t)COUT * CH * 2);
  w.ic     = (__hip_bfloat16*)alloc((size_t)32768 * 3 * CH * 2);
  w.xd     = (__hip_bfloat16*)alloc((size_t)32768 * CH * 2);
  w.qb     = (__hip_bfloat16*)alloc((size_t)RTOT * CH * 2);
  w.kb     = (__hip_bfloat16*)alloc((size_t)RTOT * CH * 2);
  w.vb     = (__hip_bfloat16*)alloc((size_t)RTOT * CH * 2);
  w.tmpf   = (float*)alloc((size_t)RTOT * CH * 4);
  w.scores = (float*)alloc((size_t)65536 * 256 * 4);
  w.pb     = (__hip_bfloat16*)alloc((size_t)65536 * 256 * 2);
  w.oc     = (__hip_bfloat16*)alloc((size_t)RTOT * CH * 2);
  w.od     = (__hip_bfloat16*)alloc((size_t)RTOT * CH * 2);
  w.os     = (__hip_bfloat16*)alloc((size_t)RTOT * CH * 2);
  w.statsR = (float*)alloc((size_t)BATCH * CH * 2 * 4);
  w.statsC = (float*)alloc((size_t)BATCH * CH * 2 * 4);
  w.comb   = (__hip_bfloat16*)alloc((size_t)RTOT * CH * 2);

  // convert all GEMM weights to bf16 once
  auto wcvt = [&](const float* src, size_t n) {
    __hip_bfloat16* d = (__hip_bfloat16*)alloc(n * 2);
    k_f2b<<<gsblocks((long long)n), 256, 0, stream>>>(src, d, (long long)n);
    return (const __hip_bfloat16*)d;
  };
  for (int i = 0; i < 9; ++i) {
    rowm.wb[i] = wcvt(rowm.p[2 + 2 * i], (size_t)CH * CH);
    colm.wb[i] = wcvt(colm.p[2 + 2 * i], (size_t)CH * CH);
  }
  k_f2b<<<gsblocks((long long)CH * CIN), 256, 0, stream>>>(in_w, w.in_wb, (long long)CH * CIN);
  k_f2b<<<gsblocks((long long)COUT * CH), 256, 0, stream>>>(out_w, w.out_wb, (long long)COUT * CH);

  // stage 0: NCHW->NHWC bf16, then in-proj GEMM with fused bias+prelu+PE
  k_nchw2nhwc<<<gsblocks((long long)RTOT * CIN), 256, 0, stream>>>(x, w.xT, (long long)RTOT * CIN);
  gemm(stream, w.xT, CIN, 0, w.in_wb, false, CIN, 0, in_b,
       w.hf, w.hb, CH, 0, RTOT, CH, CIN, 1, 2, a_in, pe);

  // prep conv weights [O,I,3] -> [O,3,I] bf16
  {
    long long n = (long long)CH * 3 * CH;
    k_prep_dsw<<<gsblocks(n), 256, 0, stream>>>(rowm.p[M_DSW], w.dswT_r, n);
    k_prep_dsw<<<gsblocks(n), 256, 0, stream>>>(colm.p[M_DSW], w.dswT_c, n);
  }
  rowm.dswT = w.dswT_r;
  colm.dswT = w.dswT_c;

  // row attention: 1024 sequences of length 64
  run_mrha(stream, rowm, w.hb, w.rowo, BATCH * D2, D1, w);

  // col attention: 256 sequences of length 256 (transposed layout)
  k_row2col<<<gsblocks((long long)RTOT * CH), 256, 0, stream>>>(w.hb, w.hcol, (long long)RTOT * CH);
  run_mrha(stream, colm, w.hcol, w.colo, BATCH * D1, D2, w);

  // instance norms + combine + prelu -> bf16
  k_instat<<<BATCH * CH, 256, 0, stream>>>(w.rowo, w.statsR);
  k_instat<<<BATCH * CH, 256, 0, stream>>>(w.colo, w.statsC);
  k_combine<<<gsblocks((long long)RTOT * CH), 256, 0, stream>>>(
      w.hf, w.rowo, w.colo, w.statsR, w.statsC, rn_g, rn_b, cn_g, cn_b,
      k1, k2, a_out, w.comb, (long long)RTOT * CH);

  // output projection, scatter to NCHW f32 output
  gemm(stream, w.comb, CH, 0, w.out_wb, false, CH, 0, out_b,
       (float*)d_out, nullptr, 0, 0, RTOT, COUT, CH, 1, 4);
}
